// S2ConvNet_original_34007551050224
// MI455X (gfx1250) — matrix operand order repaired and checked by reading, not compiled, not run on hardware
//
#include <hip/hip_runtime.h>
#include <hip/hip_bf16.h>

// ---------------------------------------------------------------------------
// S2ConvNet on MI455X (gfx1250, wave32, WMMA).
// Heavy contractions -> v_wmma_f32_16x16x32_bf16 tiled GEMMs.
// Tiles staged with GLOBAL_LOAD_ASYNC_TO_LDS_B128 (ASYNCcnt), double-buffered
// so the async DMA of tile k+1 overlaps the WMMA work on tile k.
// The -Aim plane needed by complex WMMA is materialized at the producer, so
// complex-GEMM staging is 100% async (no VGPR round-trip).
// ---------------------------------------------------------------------------

#define PI_D 3.14159265358979323846

// ---- problem constants ----
#define BATCH 64
#define F_IN  1029
#define F1    20
#define F2    40
#define NM1   19     // 2*B1-1
#define NM2   11     // 2*B2-1
#define NL1   10     // B1
#define NL2   6      // B2
#define NA1   20     // 2*B1
#define NK2   12     // 2*B2

// GEMM1: C[65856,380] = x[65856,900] * Wc[900,380]  (real, bf16 wmma)
#define G1_M   (BATCH*F_IN)   // 65856
#define G1_K   900
#define G1_KP  928
#define G1_N   380
#define G1_NP  384
// GEMM2: per-l complex, z[1216,380] = xh[1216,1029] * conj(yh)[1029,380]
#define G2_M   (BATCH*NM1)    // 1216
#define G2_KP  1056
#define G2_N   (F1*NM1)       // 380
#define G2_NP  384
// GEMM3 (z2): per-l complex, [704,440] = A[704,220] * B[220,440]
#define G3_M   (BATCH*NM2)    // 704
#define G3_K   (F1*NM2)       // 220
#define G3_KP  224
#define G3_N   (F2*NM2)       // 440
#define G3_NP  448

typedef __bf16 bf16_t;
typedef __attribute__((ext_vector_type(16))) __bf16 bf16x16;
typedef __attribute__((ext_vector_type(8)))  float  floatx8;

// ---------------------------------------------------------------------------
// small device helpers
// ---------------------------------------------------------------------------
__device__ inline unsigned short f32_to_bf16_bits(float f) {
  unsigned int u = __float_as_uint(f);
  unsigned int r = (u + 0x7FFFu + ((u >> 16) & 1u)) >> 16;
  return (unsigned short)r;
}
__device__ inline unsigned int pack2_bf16(float a, float b) {
  return (unsigned int)f32_to_bf16_bits(a) | ((unsigned int)f32_to_bf16_bits(b) << 16);
}

// 32-bit LDS byte offset of a shared-memory pointer (for async-to-LDS ops)
__device__ inline unsigned lds_addr32(const void* p) {
  return (unsigned)(unsigned long long)
      (__attribute__((address_space(3))) const char*)p;
}
// async 16B global->LDS copy, per-lane addresses; tracked by ASYNCcnt
__device__ inline void async_b128(unsigned lds_off, const void* gaddr) {
  asm volatile("global_load_async_to_lds_b128 %0, %1, off"
               :: "v"(lds_off), "v"(gaddr) : "memory");
}
__device__ inline void wait_async0() {
  asm volatile("s_wait_asynccnt 0" ::: "memory");
}

__device__ inline double dfact(int n) {
  double r = 1.0;
  for (int i = 2; i <= n; ++i) r *= (double)i;
  return r;
}

// Wigner small-d element d^l_{mp,mm}(beta); zero outside |mp|,|mm|<=l.
__device__ double wig_d(int l, int mp, int mm, double beta) {
  int amp = mp < 0 ? -mp : mp, amm = mm < 0 ? -mm : mm;
  if (amp > l || amm > l) return 0.0;
  double cb = cos(beta * 0.5), sb = sin(beta * 0.5);
  double pref = sqrt(dfact(l + mp) * dfact(l - mp) * dfact(l + mm) * dfact(l - mm));
  int s0 = (mm - mp) > 0 ? (mm - mp) : 0;
  int s1 = (l + mm) < (l - mp) ? (l + mm) : (l - mp);
  double v = 0.0;
  for (int s = s0; s <= s1; ++s) {
    double sign = ((mp - mm + s) & 1) ? -1.0 : 1.0;
    double den = dfact(l + mm - s) * dfact(s) * dfact(mp - mm + s) * dfact(l - mp - s);
    double t = sign / den;
    t *= pow(cb, (double)(2 * l + mm - mp - 2 * s));
    t *= pow(sb, (double)(mp - mm + 2 * s));
    v += t;
  }
  return pref * v;
}

__device__ double dh_weight(int b, double beta) {
  double acc = 0.0;
  for (int j = 0; j < b; ++j) acc += sin((double)(2 * j + 1) * beta) / (double)(2 * j + 1);
  return (2.0 / (double)b) * sin(beta) * acc;
}

// deterministic splitmix64 -> U[0,1): stand-in for np.default_rng(42)
__device__ double u01(unsigned long long i) {
  unsigned long long z = i + 0x9E3779B97F4A7C15ULL;
  z = (z ^ (z >> 30)) * 0xBF58476D1CE4E5B9ULL;
  z = (z ^ (z >> 27)) * 0x94D049BB133111EBULL;
  z = z ^ (z >> 31);
  return (double)(z >> 11) * (1.0 / 9007199254740992.0);
}

// ---------------------------------------------------------------------------
// zero fill
// ---------------------------------------------------------------------------
__global__ void fill_u16(unsigned short* p, long long n) {
  long long i = (long long)blockIdx.x * blockDim.x + threadIdx.x;
  long long st = (long long)gridDim.x * blockDim.x;
  for (; i < n; i += st) p[i] = 0;
}

// ---------------------------------------------------------------------------
// basis builders
// ---------------------------------------------------------------------------
// Wc column-major [n=384][k=928] bf16 : n<190 -> Re, 190..379 -> Im of
// Ea_in[a,m]*d0w[k,l,m]; zero padding elsewhere.
__global__ void build_wc(unsigned short* wc) {
  int o = blockIdx.x * blockDim.x + threadIdx.x;
  if (o >= G1_NP * G1_KP) return;
  int n = o / G1_KP, ka = o % G1_KP;
  double val = 0.0;
  if (ka < G1_K && n < G1_N) {
    int k = ka / 30, a = ka % 30;
    int plane = n / 190, lm = n % 190, l = lm / 19, mm = lm % 19 - 9;
    if (mm >= -l && mm <= l) {
      double beta = PI_D * (2.0 * k + 1.0) / 60.0;
      double alpha = 2.0 * PI_D * (double)a / 30.0;
      double d0 = wig_d(l, mm, 0, beta) * dh_weight(15, beta) / (4.0 * PI_D);
      double sc = 2.0 * PI_D / 30.0;
      val = d0 * sc * (plane == 0 ? cos(alpha * (double)mm) : -sin(alpha * (double)mm));
    }
  }
  wc[o] = f32_to_bf16_bits((float)val);
}

// K1[j=128][lm=190] complex planar f32 (deterministic stand-in S2 grid)
__global__ void build_k1(float* k1re, float* k1im) {
  int i = blockIdx.x * blockDim.x + threadIdx.x;
  if (i >= 128 * 190) return;
  int j = i / 190, lm = i % 190, l = lm / 19, mm = lm % 19 - 9;
  float re = 0.f, im = 0.f;
  int amm = mm < 0 ? -mm : mm;
  if (amm <= l) {
    double cb = 2.0 * u01((unsigned long long)j) - 1.0;
    double beta = acos(cb);
    double alpha = 2.0 * PI_D * u01(128ULL + (unsigned long long)j);
    double d0 = wig_d(l, mm, 0, beta);
    re = (float)(d0 * cos(alpha * (double)mm));
    im = (float)(-d0 * sin(alpha * (double)mm));
  }
  k1re[i] = re; k1im[i] = im;
}

// yhB column-major [l][gn=384 pad][f=1056 pad] bf16
//   = conj( sum_j kern1[f,g,j]*K1[j,l,n] )
__global__ void build_yhB(const float* __restrict__ kern1,
                          const float* __restrict__ k1re, const float* __restrict__ k1im,
                          unsigned short* yhBre, unsigned short* yhBim) {
  long long i = (long long)blockIdx.x * blockDim.x + threadIdx.x;
  const long long total = (long long)NL1 * F_IN * G2_N;
  if (i >= total) return;
  int l = (int)(i / ((long long)F_IN * G2_N));
  int rem = (int)(i % ((long long)F_IN * G2_N));
  int f = rem / G2_N, gn = rem % G2_N;
  int g = gn / 19, n = gn % 19;
  const float* kr = kern1 + ((long long)f * F1 + g) * 128;
  int base = l * 19 + n;
  float ar = 0.f, ai = 0.f;
  for (int j = 0; j < 128; ++j) {
    float kv = kr[j];
    ar += kv * k1re[j * 190 + base];
    ai += kv * k1im[j * 190 + base];
  }
  long long o = ((long long)l * G2_NP + gn) * G2_KP + f;
  yhBre[o] = f32_to_bf16_bits(ar);
  yhBim[o] = f32_to_bf16_bits(-ai);   // conj
}

__global__ void build_d4s1(float* d4s1) {  // [k20][l10][m19][n19] * (2l+1)
  int i = blockIdx.x * blockDim.x + threadIdx.x;
  if (i >= 20 * 10 * 19 * 19) return;
  int k = i / 3610, rem = i % 3610, l = rem / 361;
  int mi = (rem % 361) / 19, ni = rem % 19;
  double beta = PI_D * (2.0 * k + 1.0) / 40.0;
  d4s1[i] = (float)(wig_d(l, mi - 9, ni - 9, beta) * (double)(2 * l + 1));
}

__global__ void build_es1(float* re, float* im) {  // Es1[m19][a20] = exp(i*mm*alpha)
  int i = blockIdx.x * blockDim.x + threadIdx.x;
  if (i >= 19 * 20) return;
  int mm = i / 20 - 9, a = i % 20;
  double al = 2.0 * PI_D * (double)a / 20.0;
  re[i] = (float)cos((double)mm * al);
  im[i] = (float)sin((double)mm * al);
}

__global__ void build_ea2(float* re, float* im) {  // Ea2[a20][m11] = exp(-i*a*mm)*(2pi/20)
  int i = blockIdx.x * blockDim.x + threadIdx.x;
  if (i >= 20 * 11) return;
  int a = i / 11, mm = i % 11 - 5;
  double al = 2.0 * PI_D * (double)a / 20.0, sc = 2.0 * PI_D / 20.0;
  re[i] = (float)(sc * cos(al * (double)mm));
  im[i] = (float)(-sc * sin(al * (double)mm));
}

__global__ void build_d4a2(float* d4a2) {  // [k20][l6][m11][n11] * dh(B1)/(8pi^2)
  int i = blockIdx.x * blockDim.x + threadIdx.x;
  if (i >= 20 * 6 * 11 * 11) return;
  int k = i / 726, l = (i % 726) / 121, mi = (i % 121) / 11, ni = i % 11;
  double beta = PI_D * (2.0 * k + 1.0) / 40.0;
  d4a2[i] = (float)(wig_d(l, mi - 5, ni - 5, beta) * dh_weight(10, beta) / (8.0 * PI_D * PI_D));
}

__global__ void build_k2(float* re, float* im) {  // K2[j144][l6][n11][p11]
  int i = blockIdx.x * blockDim.x + threadIdx.x;
  if (i >= 144 * 6 * 11 * 11) return;
  int j = i / 726, l = (i % 726) / 121, mi = (i % 121) / 11, ni = i % 11;
  int ib = j / 48, ia = (j % 48) / 6, ig = j % 6;
  double beta = (double)(ib + 1) * (PI_D / 8.0) / 3.0;
  double alpha = 2.0 * PI_D * (double)ia / 8.0;
  double gamma = 2.0 * PI_D * (double)ig / 6.0;
  int mm = mi - 5, nn = ni - 5;
  double d = wig_d(l, mm, nn, beta);
  double ph = alpha * (double)mm + gamma * (double)nn;
  re[i] = (float)(d * cos(ph));
  im[i] = (float)(-d * sin(ph));
}

__global__ void build_d4s2(float* d4s2) {  // [k12][l6][m11][n11] * (2l+1)
  int i = blockIdx.x * blockDim.x + threadIdx.x;
  if (i >= 12 * 6 * 11 * 11) return;
  int k = i / 726, l = (i % 726) / 121, mi = (i % 121) / 11, ni = i % 11;
  double beta = PI_D * (2.0 * k + 1.0) / 24.0;
  d4s2[i] = (float)(wig_d(l, mi - 5, ni - 5, beta) * (double)(2 * l + 1));
}

__global__ void build_es2(float* re, float* im) {  // Es2[m11][c12] = exp(i*mm*alpha)
  int i = blockIdx.x * blockDim.x + threadIdx.x;
  if (i >= 11 * 12) return;
  int mm = i / 12 - 5, c = i % 12;
  double al = 2.0 * PI_D * (double)c / 12.0;
  re[i] = (float)cos((double)mm * al);
  im[i] = (float)sin((double)mm * al);
}

__global__ void build_wint(float* w) {  // w_int[k12]
  int i = blockIdx.x * blockDim.x + threadIdx.x;
  if (i >= 12) return;
  double beta = PI_D * (2.0 * i + 1.0) / 24.0;
  double sc = (2.0 * PI_D / 12.0); sc = sc * sc / (8.0 * PI_D * PI_D);
  w[i] = (float)(dh_weight(6, beta) * sc);
}

// B2z column-major [l][gn=448 pad][fp=224 pad] bf16 = conj(sum_j kern2*K2)
__global__ void build_B2z(const float* __restrict__ kern2,
                          const float* __restrict__ k2re, const float* __restrict__ k2im,
                          unsigned short* bre, unsigned short* bim) {
  int i = blockIdx.x * blockDim.x + threadIdx.x;
  if (i >= NL2 * G3_K * G3_N) return;
  int l = i / (G3_K * G3_N), rem = i % (G3_K * G3_N);
  int fp = rem / G3_N, gn = rem % G3_N;
  int f = fp / 11, p = fp % 11, g = gn / 11, n = gn % 11;
  const float* kr = kern2 + ((long long)f * F2 + g) * 144;
  float ar = 0.f, ai = 0.f;
  for (int j = 0; j < 144; ++j) {
    int ki = ((j * NL2 + l) * 11 + n) * 11 + p;
    float kv = kr[j];
    ar += kv * k2re[ki];
    ai += kv * k2im[ki];
  }
  long long o = ((long long)l * G3_NP + gn) * G3_KP + fp;
  bre[o] = f32_to_bf16_bits(ar);
  bim[o] = f32_to_bf16_bits(-ai);   // conj
}

// ---------------------------------------------------------------------------
// WMMA fragment helpers (16x16x32 bf16; A and B both use contiguous k = 16*(lane/16) + s)
// ---------------------------------------------------------------------------
union FragBF { bf16x16 v; unsigned int u[8]; };
#define LDSP 40   // LDS row pitch in ushorts (80B: 16B-aligned, conflict-free)

// ---------------------------------------------------------------------------
// GEMM1: real bf16 WMMA, double-buffered. A = x (f32 -> bf16 on the fly),
// B = Wc via async-to-LDS. Epilogue scatters C into xhA bf16 planes
// (re, im, -im): xhA[l][(m*64+b)][f].
// ---------------------------------------------------------------------------
__global__ __launch_bounds__(256)
void sgemm1_x(const float* __restrict__ X, const unsigned short* __restrict__ Wc,
              unsigned short* __restrict__ xhAre, unsigned short* __restrict__ xhAim,
              unsigned short* __restrict__ xhAimN) {
  __shared__ __align__(16) unsigned short lA[2][64][LDSP];
  __shared__ __align__(16) unsigned short lB[2][64][LDSP];
  const int tid = threadIdx.x, lane = tid & 31, wid = tid >> 5;
  const int waveM = wid & 3, waveN = wid >> 2;
  const int bm = blockIdx.x * 64, bn = blockIdx.y * 64;
  const int lhalf = lane >> 4, lrow = lane & 15;
  const int sr = tid >> 2, sc = (tid & 3) * 8;   // one 16B chunk per thread

  floatx8 acc[2];
  for (int t = 0; t < 2; ++t) for (int r = 0; r < 8; ++r) acc[t][r] = 0.f;

  auto stage = [&](int buf, int kk) {
    // B asynchronously (column-major [N][K], fully padded)
    async_b128(lds_addr32(&lB[buf][sr][sc]),
               Wc + (long long)(bn + sr) * G1_KP + kk + sc);
    // A: 8 f32 -> 8 bf16 (uniform branch for the single partial step)
    if (kk != 896) {
      const float* xp = X + (long long)(bm + sr) * G1_K + kk + sc;
      float4 a = *(const float4*)xp;
      float4 b = *(const float4*)(xp + 4);
      uint4 q;
      q.x = pack2_bf16(a.x, a.y); q.y = pack2_bf16(a.z, a.w);
      q.z = pack2_bf16(b.x, b.y); q.w = pack2_bf16(b.z, b.w);
      *(uint4*)&lA[buf][sr][sc] = q;
    } else {
      float v[8];
#pragma unroll
      for (int e = 0; e < 8; ++e) {
        int k = kk + sc + e;
        v[e] = (k < G1_K) ? X[(long long)(bm + sr) * G1_K + k] : 0.f;
      }
      uint4 q;
      q.x = pack2_bf16(v[0], v[1]); q.y = pack2_bf16(v[2], v[3]);
      q.z = pack2_bf16(v[4], v[5]); q.w = pack2_bf16(v[6], v[7]);
      *(uint4*)&lA[buf][sr][sc] = q;
    }
  };

  stage(0, 0);
  int ib = 0;
  for (int kk = 0; kk < G1_KP; kk += 32, ib ^= 1) {
    wait_async0();
    __syncthreads();
    if (kk + 32 < G1_KP) stage(ib ^ 1, kk + 32);  // overlap with compute

    FragBF fa;
    int arow = waveM * 16 + lrow;
#pragma unroll
    for (int v = 0; v < 8; ++v) {
      int kof = lhalf * 16 + 2 * v;
      fa.u[v] = *(const unsigned int*)&lA[ib][arow][kof];
    }
#pragma unroll
    for (int t = 0; t < 2; ++t) {
      FragBF fb;
      int bcol = waveN * 32 + t * 16 + lrow;
#pragma unroll
      for (int v = 0; v < 8; ++v) {
        int kof = lhalf * 16 + 2 * v;
        fb.u[v] = *(const unsigned int*)&lB[ib][bcol][kof];
      }
      acc[t] = __builtin_amdgcn_wmma_f32_16x16x32_bf16(
          false, fa.v, false, fb.v, (short)0, acc[t], false, false);
    }
  }
  // scatter epilogue: row=(b,f), col -> (plane, l, m); im also negated plane
#pragma unroll
  for (int t = 0; t < 2; ++t)
#pragma unroll
    for (int r = 0; r < 8; ++r) {
      int row = bm + waveM * 16 + lhalf * 8 + r;
      int col = bn + waveN * 32 + t * 16 + lrow;
      if (col < G1_N) {
        int b = row / F_IN, f = row - b * F_IN;
        int plane = col / 190, lm = col - plane * 190;
        int l = lm / 19, m = lm - l * 19;
        long long idx = ((long long)(l * G2_M + m * 64 + b)) * G2_KP + f;
        unsigned short bits = f32_to_bf16_bits(acc[t][r]);
        if (plane == 0) {
          xhAre[idx] = bits;
        } else {
          xhAim[idx]  = bits;
          xhAimN[idx] = bits ^ 0x8000u;
        }
      }
    }
}

// ---------------------------------------------------------------------------
// Batched complex bf16 GEMM: C = A * B. A planes (re, im, -im) row-major
// [M][KP]; B planes (re, im) column-major [NP][KP]; C f32 planar [M][N].
// blockIdx.z = l-slice. All 5 LDS tiles staged via async-to-LDS b128,
// double-buffered (async DMA of tile k+1 overlaps WMMAs on tile k).
// ---------------------------------------------------------------------------
__global__ __launch_bounds__(256)
void cgemm_bf16(const unsigned short* __restrict__ Are, const unsigned short* __restrict__ Aim,
                const unsigned short* __restrict__ AimN,
                const unsigned short* __restrict__ Bre, const unsigned short* __restrict__ Bim,
                float* __restrict__ Cre, float* __restrict__ Cim,
                int M, int N, int KP,
                long long sA, long long sB, long long sC) {
  __shared__ __align__(16) unsigned short lAre[2][64][LDSP];
  __shared__ __align__(16) unsigned short lAim[2][64][LDSP];
  __shared__ __align__(16) unsigned short lAimN[2][64][LDSP];
  __shared__ __align__(16) unsigned short lBre[2][64][LDSP];
  __shared__ __align__(16) unsigned short lBim[2][64][LDSP];

  const int tid = threadIdx.x, lane = tid & 31, wid = tid >> 5;
  const int waveM = wid & 3, waveN = wid >> 2;
  const int bm = blockIdx.x * 64, bn = blockIdx.y * 64;
  const int lz = blockIdx.z;
  const int lhalf = lane >> 4, lrow = lane & 15;
  const int sr = tid >> 2, sc = (tid & 3) * 8;

  Are  += (long long)lz * sA; Aim += (long long)lz * sA; AimN += (long long)lz * sA;
  Bre  += (long long)lz * sB; Bim += (long long)lz * sB;
  Cre  += (long long)lz * sC; Cim += (long long)lz * sC;

  floatx8 cre[2], cim[2];
  for (int t = 0; t < 2; ++t)
    for (int r = 0; r < 8; ++r) { cre[t][r] = 0.f; cim[t][r] = 0.f; }

  auto stage = [&](int buf, int kk) {
    long long ga = (long long)(bm + sr) * KP + kk + sc;
    long long gb = (long long)(bn + sr) * KP + kk + sc;
    async_b128(lds_addr32(&lAre[buf][sr][sc]),  Are  + ga);
    async_b128(lds_addr32(&lAim[buf][sr][sc]),  Aim  + ga);
    async_b128(lds_addr32(&lAimN[buf][sr][sc]), AimN + ga);
    async_b128(lds_addr32(&lBre[buf][sr][sc]),  Bre  + gb);
    async_b128(lds_addr32(&lBim[buf][sr][sc]),  Bim  + gb);
  };

  stage(0, 0);
  int ib = 0;
  for (int kk = 0; kk < KP; kk += 32, ib ^= 1) {
    wait_async0();
    __syncthreads();
    if (kk + 32 < KP) stage(ib ^ 1, kk + 32);  // overlap with compute

    FragBF far, fai, fan;
    int arow = waveM * 16 + lrow;
#pragma unroll
    for (int v = 0; v < 8; ++v) {
      int kof = lhalf * 16 + 2 * v;
      far.u[v] = *(const unsigned int*)&lAre[ib][arow][kof];
      fai.u[v] = *(const unsigned int*)&lAim[ib][arow][kof];
      fan.u[v] = *(const unsigned int*)&lAimN[ib][arow][kof];
    }
#pragma unroll
    for (int t = 0; t < 2; ++t) {
      FragBF fbr, fbi;
      int bcol = waveN * 32 + t * 16 + lrow;
#pragma unroll
      for (int v = 0; v < 8; ++v) {
        int kof = lhalf * 16 + 2 * v;
        fbr.u[v] = *(const unsigned int*)&lBre[ib][bcol][kof];
        fbi.u[v] = *(const unsigned int*)&lBim[ib][bcol][kof];
      }
      // Cre += Are*Bre - Aim*Bim ; Cim += Are*Bim + Aim*Bre
      cre[t] = __builtin_amdgcn_wmma_f32_16x16x32_bf16(false, far.v, false, fbr.v, (short)0, cre[t], false, false);
      cre[t] = __builtin_amdgcn_wmma_f32_16x16x32_bf16(false, fan.v, false, fbi.v, (short)0, cre[t], false, false);
      cim[t] = __builtin_amdgcn_wmma_f32_16x16x32_bf16(false, far.v, false, fbi.v, (short)0, cim[t], false, false);
      cim[t] = __builtin_amdgcn_wmma_f32_16x16x32_bf16(false, fai.v, false, fbr.v, (short)0, cim[t], false, false);
    }
  }
#pragma unroll
  for (int t = 0; t < 2; ++t)
#pragma unroll
    for (int r = 0; r < 8; ++r) {
      int row = bm + waveM * 16 + lhalf * 8 + r;
      int col = bn + waveN * 32 + t * 16 + lrow;
      if (row < M && col < N) {
        long long ci = (long long)row * N + col;
        Cre[ci] = cre[t][r]; Cim[ci] = cim[t][r];
      }
    }
}

// ---------------------------------------------------------------------------
// VALU einsum stages
// ---------------------------------------------------------------------------
// s[b,g,k,m,n] = sum_l z[l][(m,b)][(g,n)] * d4s1[k,l,m,n]
__global__ void s_stage(const float* __restrict__ zre, const float* __restrict__ zim,
                        const float* __restrict__ d4s1, float* sre, float* sim) {
  long long i = (long long)blockIdx.x * blockDim.x + threadIdx.x;
  const long long total = (long long)BATCH * F1 * NA1 * NM1 * NM1;
  if (i >= total) return;
  int n = (int)(i % 19), m = (int)((i / 19) % 19), k = (int)((i / 361) % 20);
  int g = (int)((i / 7220) % 20), b = (int)(i / 144400);
  float ar = 0.f, ai = 0.f;
  for (int l = 0; l < NL1; ++l) {
    float d = d4s1[((k * 10 + l) * 19 + m) * 19 + n];
    long long zi = ((long long)(l * G2_M + m * 64 + b)) * G2_N + g * 19 + n;
    ar += zre[zi] * d; ai += zim[zi] * d;
  }
  sre[i] = ar; sim[i] = ai;
}

// u[b,g,k,a,n] = sum_m s[b,g,k,m,n] * Es1[m,a]
__global__ void u_stage(const float* __restrict__ sre, const float* __restrict__ sim,
                        const float* __restrict__ e1r, const float* __restrict__ e1i,
                        float* ure, float* uim) {
  long long i = (long long)blockIdx.x * blockDim.x + threadIdx.x;
  const long long total = (long long)BATCH * F1 * NA1 * NA1 * NM1;
  if (i >= total) return;
  int n = (int)(i % 19), a = (int)((i / 19) % 20), k = (int)((i / 380) % 20);
  int g = (int)((i / 7600) % 20), b = (int)(i / 152000);
  float ar = 0.f, ai = 0.f;
  long long sb = (((long long)(b * 20 + g) * 20 + k) * 19) * 19 + n;
  for (int m = 0; m < 19; ++m) {
    float sr = sre[sb + (long long)m * 19], si = sim[sb + (long long)m * 19];
    float er = e1r[m * 20 + a], ei = e1i[m * 20 + a];
    ar += sr * er - si * ei;
    ai += sr * ei + si * er;
  }
  ure[i] = ar; uim[i] = ai;
}

// x1[b,g,k,a,c] = relu( Re(sum_n u * Es1[n,c]) + bias1[g] )
__global__ void x1_stage(const float* __restrict__ ure, const float* __restrict__ uim,
                         const float* __restrict__ e1r, const float* __restrict__ e1i,
                         const float* __restrict__ bias1, float* x1) {
  long long i = (long long)blockIdx.x * blockDim.x + threadIdx.x;
  const long long total = (long long)BATCH * F1 * NA1 * NA1 * NA1;
  if (i >= total) return;
  int c = (int)(i % 20), a = (int)((i / 20) % 20), k = (int)((i / 400) % 20);
  int g = (int)((i / 8000) % 20), b = (int)(i / 160000);
  float acc = 0.f;
  long long ub = ((((long long)(b * 20 + g) * 20 + k) * 20 + a) * 19);
  for (int n = 0; n < 19; ++n)
    acc += ure[ub + n] * e1r[n * 20 + c] - uim[ub + n] * e1i[n * 20 + c];
  acc += bias1[g];
  x1[i] = acc > 0.f ? acc : 0.f;
}

// t1[b,f,k,m,c] = sum_a x1[b,f,k,a,c] * Ea2[a,m]
__global__ void t1_stage(const float* __restrict__ x1,
                         const float* __restrict__ e2r, const float* __restrict__ e2i,
                         float* t1r, float* t1i) {
  long long i = (long long)blockIdx.x * blockDim.x + threadIdx.x;
  const long long total = (long long)BATCH * F1 * NA1 * NM2 * NA1;
  if (i >= total) return;
  int c = (int)(i % 20), m = (int)((i / 20) % 11), k = (int)((i / 220) % 20);
  int f = (int)((i / 4400) % 20), b = (int)(i / 88000);
  float ar = 0.f, ai = 0.f;
  long long xb = (((long long)(b * 20 + f) * 20 + k) * 20) * 20 + c;
  for (int a = 0; a < 20; ++a) {
    float xv = x1[xb + (long long)a * 20];
    ar += xv * e2r[a * 11 + m];
    ai += xv * e2i[a * 11 + m];
  }
  t1r[i] = ar; t1i[i] = ai;
}

// t2[b,f,k,m,n] = sum_c t1[b,f,k,m,c] * Ea2[c,n]
__global__ void t2_stage(const float* __restrict__ t1r, const float* __restrict__ t1i,
                         const float* __restrict__ e2r, const float* __restrict__ e2i,
                         float* t2r, float* t2i) {
  long long i = (long long)blockIdx.x * blockDim.x + threadIdx.x;
  const long long total = (long long)BATCH * F1 * NA1 * NM2 * NM2;
  if (i >= total) return;
  int n = (int)(i % 11), m = (int)((i / 11) % 11), k = (int)((i / 121) % 20);
  int f = (int)((i / 2420) % 20), b = (int)(i / 48400);
  float ar = 0.f, ai = 0.f;
  long long tb = ((((long long)(b * 20 + f) * 20 + k) * 11 + m) * 20);
  for (int c = 0; c < 20; ++c) {
    float pr = t1r[tb + c], pi = t1i[tb + c];
    float er = e2r[c * 11 + n], ei = e2i[c * 11 + n];
    ar += pr * er - pi * ei;
    ai += pr * ei + pi * er;
  }
  t2r[i] = ar; t2i[i] = ai;
}

// xh2[b,f,l,m,n] = sum_k t2[b,f,k,m,n] * d4a2[k,l,m,n]
__global__ void xh2_stage(const float* __restrict__ t2r, const float* __restrict__ t2i,
                          const float* __restrict__ d4a2, float* xr, float* xi) {
  long long i = (long long)blockIdx.x * blockDim.x + threadIdx.x;
  const long long total = (long long)BATCH * F1 * NL2 * NM2 * NM2;
  if (i >= total) return;
  int n = (int)(i % 11), m = (int)((i / 11) % 11), l = (int)((i / 121) % 6);
  int f = (int)((i / 726) % 20), b = (int)(i / 14520);
  float ar = 0.f, ai = 0.f;
  for (int k = 0; k < 20; ++k) {
    float d = d4a2[((k * 6 + l) * 11 + m) * 11 + n];
    long long ti = (((long long)(b * 20 + f) * 20 + k) * 11 + m) * 11 + n;
    ar += t2r[ti] * d; ai += t2i[ti] * d;
  }
  xr[i] = ar; xi[i] = ai;
}

// A2z[l][(m,b)][fp] bf16 planes (re, im, -im) <- xh2[b,f,l,m,p]
__global__ void repackA2(const float* __restrict__ xr, const float* __restrict__ xi,
                         unsigned short* are, unsigned short* aim, unsigned short* aimn) {
  long long i = (long long)blockIdx.x * blockDim.x + threadIdx.x;
  const long long total = (long long)NL2 * G3_M * G3_K;
  if (i >= total) return;
  int l = (int)(i / ((long long)G3_M * G3_K));
  int rem = (int)(i % ((long long)G3_M * G3_K));
  int r = rem / G3_K, fp = rem % G3_K;
  int m = r / 64, b = r % 64, f = fp / 11, p = fp % 11;
  long long si = (((long long)(b * 20 + f) * 6 + l) * 11 + m) * 11 + p;
  long long o = ((long long)l * G3_M + r) * G3_KP + fp;
  unsigned short bi = f32_to_bf16_bits(xi[si]);
  are[o]  = f32_to_bf16_bits(xr[si]);
  aim[o]  = bi;
  aimn[o] = bi ^ 0x8000u;
}

// s2[b,g,k,m,n] = sum_l z2[l][(m,b)][(g,n)] * d4s2[k,l,m,n]
__global__ void s2_stage(const float* __restrict__ zr, const float* __restrict__ zi,
                         const float* __restrict__ d4s2, float* sr, float* si) {
  long long i = (long long)blockIdx.x * blockDim.x + threadIdx.x;
  const long long total = (long long)BATCH * F2 * NK2 * NM2 * NM2;
  if (i >= total) return;
  int n = (int)(i % 11), m = (int)((i / 11) % 11), k = (int)((i / 121) % 12);
  int g = (int)((i / 1452) % 40), b = (int)(i / 58080);
  float ar = 0.f, ai = 0.f;
  for (int l = 0; l < NL2; ++l) {
    float d = d4s2[((k * 6 + l) * 11 + m) * 11 + n];
    long long zidx = ((long long)(l * G3_M + m * 64 + b)) * G3_N + g * 11 + n;
    ar += zr[zidx] * d; ai += zi[zidx] * d;
  }
  sr[i] = ar; si[i] = ai;
}

// u2[b,g,k,a,n] = sum_m s2 * Es2[m,a]
__global__ void u2_stage(const float* __restrict__ sr, const float* __restrict__ si,
                         const float* __restrict__ er, const float* __restrict__ ei,
                         float* ur, float* ui) {
  long long i = (long long)blockIdx.x * blockDim.x + threadIdx.x;
  const long long total = (long long)BATCH * F2 * NK2 * NK2 * NM2;
  if (i >= total) return;
  int n = (int)(i % 11), a = (int)((i / 11) % 12), k = (int)((i / 132) % 12);
  int g = (int)((i / 1584) % 40), b = (int)(i / 63360);
  float ar = 0.f, ai = 0.f;
  long long sb = (((long long)(b * 40 + g) * 12 + k) * 11) * 11 + n;
  for (int m = 0; m < 11; ++m) {
    float pr = sr[sb + (long long)m * 11], pi = si[sb + (long long)m * 11];
    float cr = er[m * 12 + a], ci = ei[m * 12 + a];
    ar += pr * cr - pi * ci;
    ai += pr * ci + pi * cr;
  }
  ur[i] = ar; ui[i] = ai;
}

// x2[b,g,k,a,c] = relu( Re(sum_n u2 * Es2[n,c]) + bias2[g] )
__global__ void x2_stage(const float* __restrict__ ur, const float* __restrict__ ui,
                         const float* __restrict__ er, const float* __restrict__ ei,
                         const float* __restrict__ bias2, float* x2) {
  long long i = (long long)blockIdx.x * blockDim.x + threadIdx.x;
  const long long total = (long long)BATCH * F2 * NK2 * NK2 * NK2;
  if (i >= total) return;
  int c = (int)(i % 12), a = (int)((i / 12) % 12), k = (int)((i / 144) % 12);
  int g = (int)((i / 1728) % 40), b = (int)(i / 69120);
  float acc = 0.f;
  long long ub = ((((long long)(b * 40 + g) * 12 + k) * 12 + a) * 11);
  for (int n = 0; n < 11; ++n)
    acc += ur[ub + n] * er[n * 12 + c] - ui[ub + n] * ei[n * 12 + c];
  acc += bias2[g];
  x2[i] = acc > 0.f ? acc : 0.f;
}

// head: feat[b,g] = sum_{k,a,c} x2*w_int[k]; out = feat @ W.T + b_lin
__global__ void head_stage(const float* __restrict__ x2, const float* __restrict__ wint,
                           const float* __restrict__ W, const float* __restrict__ blin,
                           float* out) {
  __shared__ float feat[F2];
  const int tid = threadIdx.x, b = blockIdx.x;
  if (tid < F2) feat[tid] = 0.f;
  __syncthreads();
  for (int i = tid; i < F2 * 1728; i += blockDim.x) {
    int g = i / 1728, rem = i % 1728, k = rem / 144;
    float v = x2[(long long)(b * F2 + g) * 1728 + rem] * wint[k];
    atomicAdd(&feat[g], v);
  }
  __syncthreads();
  if (tid < 2) {
    float o = blin[tid];
    for (int g = 0; g < F2; ++g) o += feat[g] * W[tid * F2 + g];
    out[b * 2 + tid] = o;
  }
}

// ---------------------------------------------------------------------------
// host-side orchestration
// ---------------------------------------------------------------------------
static inline int nblk(long long n, int t) { return (int)((n + t - 1) / t); }

extern "C" void kernel_launch(void* const* d_in, const int* in_sizes, int n_in,
                              void* d_out, int out_size, void* d_ws, size_t ws_size,
                              hipStream_t stream) {
  const float* x     = (const float*)d_in[0];
  const float* kern1 = (const float*)d_in[1];
  const float* bias1 = (const float*)d_in[2];
  const float* kern2 = (const float*)d_in[3];
  const float* bias2 = (const float*)d_in[4];
  const float* Wlin  = (const float*)d_in[5];
  const float* blin  = (const float*)d_in[6];
  float* out = (float*)d_out;

  char* ws = (char*)d_ws;
  size_t off = 0;
  auto bump = [&](size_t sz) -> char* {
    char* p = ws + off;
    off = (off + sz + 255) & ~(size_t)255;
    return p;
  };

  // persistent basis / operands
  unsigned short* wc    = (unsigned short*)bump((size_t)G1_NP * G1_KP * 2);
  float* k1re = (float*)bump(128 * 190 * 4);
  float* k1im = (float*)bump(128 * 190 * 4);
  unsigned short* yhBre = (unsigned short*)bump((size_t)NL1 * G2_NP * G2_KP * 2);
  unsigned short* yhBim = (unsigned short*)bump((size_t)NL1 * G2_NP * G2_KP * 2);
  float* d4s1 = (float*)bump(20 * 10 * 19 * 19 * 4);
  float* e1r  = (float*)bump(19 * 20 * 4);
  float* e1i  = (float*)bump(19 * 20 * 4);
  float* d4a2 = (float*)bump(20 * 6 * 11 * 11 * 4);
  float* e2r  = (float*)bump(20 * 11 * 4);
  float* e2i  = (float*)bump(20 * 11 * 4);
  float* k2re = (float*)bump(144 * 6 * 11 * 11 * 4);
  float* k2im = (float*)bump(144 * 6 * 11 * 11 * 4);
  float* d4s2 = (float*)bump(12 * 6 * 11 * 11 * 4);
  float* e3r  = (float*)bump(11 * 12 * 4);
  float* e3i  = (float*)bump(11 * 12 * 4);
  float* wint = (float*)bump(12 * 4);
  unsigned short* B2zre = (unsigned short*)bump((size_t)NL2 * G3_NP * G3_KP * 2);
  unsigned short* B2zim = (unsigned short*)bump((size_t)NL2 * G3_NP * G3_KP * 2);

  // ping-pong arenas (80 MB each)
  const size_t ARENA = (size_t)80 << 20;
  char* arA = bump(ARENA);
  char* arB = bump(ARENA);

  // arena A views
  unsigned short* xhAre = (unsigned short*)arA;
  unsigned short* xhAim  = xhAre + (size_t)NL1 * G2_M * G2_KP;
  unsigned short* xhAimN = xhAim + (size_t)NL1 * G2_M * G2_KP;
  float* sre = (float*)arA;
  float* sim = sre + (size_t)BATCH * F1 * NA1 * NM1 * NM1;
  float* x1  = (float*)arA;
  float* t2r = (float*)arA;
  float* t2i = t2r + (size_t)BATCH * F1 * NA1 * NM2 * NM2;
  unsigned short* A2zre  = (unsigned short*)arA;
  unsigned short* A2zim  = A2zre + (size_t)NL2 * G3_M * G3_KP;
  unsigned short* A2zimN = A2zim + (size_t)NL2 * G3_M * G3_KP;
  float* s2r = (float*)arA;
  float* s2i = s2r + (size_t)BATCH * F2 * NK2 * NM2 * NM2;
  float* x2  = (float*)arA;
  // arena B views
  float* zre = (float*)arB;
  float* zim = zre + (size_t)NL1 * G2_M * G2_N;
  float* ure = (float*)arB;
  float* uim = ure + (size_t)BATCH * F1 * NA1 * NA1 * NM1;
  float* t1r = (float*)arB;
  float* t1i = t1r + (size_t)BATCH * F1 * NA1 * NM2 * NA1;
  float* xh2r = (float*)arB;
  float* xh2i = xh2r + (size_t)BATCH * F1 * NL2 * NM2 * NM2;
  float* z2r = (float*)arB;
  float* z2i = z2r + (size_t)NL2 * G3_M * G3_N;
  float* u2r = (float*)arB;
  float* u2i = u2r + (size_t)BATCH * F2 * NK2 * NK2 * NM2;

  const int T = 256;

  // ---- basis ----
  build_wc<<<nblk((long long)G1_NP * G1_KP, T), T, 0, stream>>>(wc);
  build_k1<<<nblk(128 * 190, T), T, 0, stream>>>(k1re, k1im);
  fill_u16<<<2048, T, 0, stream>>>(yhBre, (long long)NL1 * G2_NP * G2_KP);
  fill_u16<<<2048, T, 0, stream>>>(yhBim, (long long)NL1 * G2_NP * G2_KP);
  build_yhB<<<nblk((long long)NL1 * F_IN * G2_N, T), T, 0, stream>>>(kern1, k1re, k1im, yhBre, yhBim);
  build_d4s1<<<nblk(20 * 10 * 19 * 19, T), T, 0, stream>>>(d4s1);
  build_es1<<<nblk(19 * 20, T), T, 0, stream>>>(e1r, e1i);
  build_ea2<<<nblk(20 * 11, T), T, 0, stream>>>(e2r, e2i);
  build_d4a2<<<nblk(20 * 6 * 11 * 11, T), T, 0, stream>>>(d4a2);
  build_k2<<<nblk(144 * 6 * 11 * 11, T), T, 0, stream>>>(k2re, k2im);
  build_d4s2<<<nblk(12 * 6 * 11 * 11, T), T, 0, stream>>>(d4s2);
  build_es2<<<nblk(11 * 12, T), T, 0, stream>>>(e3r, e3i);
  build_wint<<<1, 32, 0, stream>>>(wint);
  fill_u16<<<2048, T, 0, stream>>>(B2zre, (long long)NL2 * G3_NP * G3_KP);
  fill_u16<<<2048, T, 0, stream>>>(B2zim, (long long)NL2 * G3_NP * G3_KP);
  build_B2z<<<nblk((long long)NL2 * G3_K * G3_N, T), T, 0, stream>>>(kern2, k2re, k2im, B2zre, B2zim);

  // ---- layer 1 ----
  fill_u16<<<2048, T, 0, stream>>>(xhAre,  (long long)NL1 * G2_M * G2_KP);
  fill_u16<<<2048, T, 0, stream>>>(xhAim,  (long long)NL1 * G2_M * G2_KP);
  fill_u16<<<2048, T, 0, stream>>>(xhAimN, (long long)NL1 * G2_M * G2_KP);
  sgemm1_x<<<dim3(G1_M / 64, G1_NP / 64), T, 0, stream>>>(x, wc, xhAre, xhAim, xhAimN);
  cgemm_bf16<<<dim3(G2_M / 64, G2_NP / 64, NL1), T, 0, stream>>>(
      xhAre, xhAim, xhAimN, yhBre, yhBim, zre, zim, G2_M, G2_N, G2_KP,
      (long long)G2_M * G2_KP, (long long)G2_NP * G2_KP, (long long)G2_M * G2_N);
  s_stage<<<nblk((long long)BATCH * F1 * NA1 * NM1 * NM1, T), T, 0, stream>>>(zre, zim, d4s1, sre, sim);
  u_stage<<<nblk((long long)BATCH * F1 * NA1 * NA1 * NM1, T), T, 0, stream>>>(sre, sim, e1r, e1i, ure, uim);
  x1_stage<<<nblk((long long)BATCH * F1 * NA1 * NA1 * NA1, T), T, 0, stream>>>(ure, uim, e1r, e1i, bias1, x1);

  // ---- layer 2 ----
  t1_stage<<<nblk((long long)BATCH * F1 * NA1 * NM2 * NA1, T), T, 0, stream>>>(x1, e2r, e2i, t1r, t1i);
  t2_stage<<<nblk((long long)BATCH * F1 * NA1 * NM2 * NM2, T), T, 0, stream>>>(t1r, t1i, e2r, e2i, t2r, t2i);
  xh2_stage<<<nblk((long long)BATCH * F1 * NL2 * NM2 * NM2, T), T, 0, stream>>>(t2r, t2i, d4a2, xh2r, xh2i);
  fill_u16<<<2048, T, 0, stream>>>(A2zre,  (long long)NL2 * G3_M * G3_KP);
  fill_u16<<<2048, T, 0, stream>>>(A2zim,  (long long)NL2 * G3_M * G3_KP);
  fill_u16<<<2048, T, 0, stream>>>(A2zimN, (long long)NL2 * G3_M * G3_KP);
  repackA2<<<nblk((long long)NL2 * G3_M * G3_K, T), T, 0, stream>>>(xh2r, xh2i, A2zre, A2zim, A2zimN);
  cgemm_bf16<<<dim3(G3_M / 64, G3_NP / 64, NL2), T, 0, stream>>>(
      A2zre, A2zim, A2zimN, B2zre, B2zim, z2r, z2i, G3_M, G3_N, G3_KP,
      (long long)G3_M * G3_KP, (long long)G3_NP * G3_KP, (long long)G3_M * G3_N);
  s2_stage<<<nblk((long long)BATCH * F2 * NK2 * NM2 * NM2, T), T, 0, stream>>>(z2r, z2i, d4s2, s2r, s2i);
  u2_stage<<<nblk((long long)BATCH * F2 * NK2 * NK2 * NM2, T), T, 0, stream>>>(s2r, s2i, e3r, e3i, u2r, u2i);
  x2_stage<<<nblk((long long)BATCH * F2 * NK2 * NK2 * NK2, T), T, 0, stream>>>(u2r, u2i, e3r, e3i, bias2, x2);

  // ---- head ----
  head_stage<<<BATCH, T, 0, stream>>>(x2, wint, Wlin, blin, out);

  (void)in_sizes; (void)n_in; (void)out_size; (void)ws_size;
}
